// NonLocalAttention_9560597201276
// MI455X (gfx1250) — compile-verified, matrix operand order checked by reading, not run on hardware
//
#include <hip/hip_runtime.h>
#include <hip/hip_bf16.h>

typedef __attribute__((ext_vector_type(16))) _Float16 vh16;
typedef __attribute__((ext_vector_type(8)))  _Float16 vh8;
typedef __attribute__((ext_vector_type(8)))  float    vf8;
typedef __attribute__((ext_vector_type(4)))  float    vf4;

#define C_CH   512
#define NSEQ   4096
#define NBATCH 4
#define GROUPS 32
#define CPG    16
#define SB_LD  520   // padded LDS row stride (halves): 1040 B -> +4 banks/row, conflict-free b128 frags

// ---------------------------------------------------------------------------
// WMMA helpers (CDNA5 16x16x32 f16 -> f32)
// ---------------------------------------------------------------------------
__device__ __forceinline__ vf8 wmma_acc(vh16 a, vh16 b, vf8 c) {
    return __builtin_amdgcn_wmma_f32_16x16x32_f16(false, a, false, b, (short)0, c,
                                                  false, false);
}

// 16-bit A/B fragment for 16x16x32. lanes 0-15: K = k0+[0..7], k0+[16..23];
// lanes 16-31: K = k0+[8..15], k0+[24..31]. rowp = base of this lane's row.
__device__ __forceinline__ vh16 frag_load(const _Float16* rowp, int k0, int lane) {
    const _Float16* p = rowp + k0 + ((lane >> 4) & 1) * 8;
    vh8 lo = *(const vh8*)(p);
    vh8 hi = *(const vh8*)(p + 16);
    return __builtin_shufflevector(lo, hi, 0,1,2,3,4,5,6,7,8,9,10,11,12,13,14,15);
}

// ---------------------------------------------------------------------------
// K0: f32 -> f16 weight conversion
// ---------------------------------------------------------------------------
__global__ __launch_bounds__(256) void cvt_f16_kernel(const float* __restrict__ in,
                                                      _Float16* __restrict__ out, int n) {
    int i = (blockIdx.x * 256 + threadIdx.x) * 4;
    if (i + 3 < n) {
        vf4 v = *(const vf4*)(in + i);
        out[i + 0] = (_Float16)v[0];
        out[i + 1] = (_Float16)v[1];
        out[i + 2] = (_Float16)v[2];
        out[i + 3] = (_Float16)v[3];
    }
}

// ---------------------------------------------------------------------------
// K1: GroupNorm. One block per (batch, group): 16 ch x 4096 = 65536 contiguous f32.
// ---------------------------------------------------------------------------
__global__ __launch_bounds__(256) void gn_kernel(const float* __restrict__ x,
                                                 const float* __restrict__ gamma,
                                                 const float* __restrict__ beta,
                                                 float* __restrict__ xn) {
    int bg = blockIdx.x;
    int g = bg & (GROUPS - 1);
    int b = bg >> 5;
    size_t base = ((size_t)b * C_CH + (size_t)g * CPG) * NSEQ;
    const vf4* src = (const vf4*)(x + base);
    vf4* dst = (vf4*)(xn + base);
    int t = threadIdx.x;

    float s = 0.0f, s2 = 0.0f;
    for (int i = 0; i < 64; ++i) {
        vf4 v = src[i * 256 + t];
        s  += v[0] + v[1] + v[2] + v[3];
        s2 += v[0]*v[0] + v[1]*v[1] + v[2]*v[2] + v[3]*v[3];
    }
    __shared__ float r1[256], r2[256];
    r1[t] = s; r2[t] = s2;
    __syncthreads();
    for (int off = 128; off > 0; off >>= 1) {
        if (t < off) { r1[t] += r1[t + off]; r2[t] += r2[t + off]; }
        __syncthreads();
    }
    float mean = r1[0] * (1.0f / 65536.0f);
    float var  = r2[0] * (1.0f / 65536.0f) - mean * mean;
    float rstd = rsqrtf(var + 1e-6f);

    for (int i = 0; i < 64; ++i) {
        int idx4 = i * 256 + t;
        int c_local = (idx4 * 4) >> 12;  // 4096 elems per channel
        float ga = gamma[g * CPG + c_local];
        float be = beta[g * CPG + c_local];
        vf4 v = src[idx4];
        vf4 o;
        o[0] = (v[0] - mean) * rstd * ga + be;
        o[1] = (v[1] - mean) * rstd * ga + be;
        o[2] = (v[2] - mean) * rstd * ga + be;
        o[3] = (v[3] - mean) * rstd * ga + be;
        dst[idx4] = o;
    }
}

// ---------------------------------------------------------------------------
// K2: transpose + convert: xn f32 [B,C,N] -> xnT f16 [B,N,C]
// ---------------------------------------------------------------------------
__global__ __launch_bounds__(256) void transpose_cvt_kernel(const float* __restrict__ xn,
                                                            _Float16* __restrict__ xnT) {
    __shared__ float tile[32][33];
    int b = blockIdx.z;
    int n0 = blockIdx.x * 32;
    int c0 = blockIdx.y * 32;
    const float* src = xn + (size_t)b * C_CH * NSEQ;
    _Float16* dst = xnT + (size_t)b * NSEQ * C_CH;
    int tx = threadIdx.x & 31;
    int ty = threadIdx.x >> 5;
    for (int r = 0; r < 4; ++r)
        tile[ty + 8 * r][tx] = src[(size_t)(c0 + ty + 8 * r) * NSEQ + n0 + tx];
    __syncthreads();
    for (int r = 0; r < 4; ++r)
        dst[(size_t)(n0 + ty + 8 * r) * C_CH + c0 + tx] = (_Float16)tile[tx][ty + 8 * r];
}

// ---------------------------------------------------------------------------
// K3: WMMA GEMM, 2x2 register blocking + async-staged B slab in LDS.
// D[m,n] = sum_k A[m,k]*B[n,k]; out stored "natural transposed": out[n*ldo+m].
// Block = 8 waves; all waves share a 32-wide n slab (B rows n_base..n_base+31,
// full K=512) staged once via GLOBAL_LOAD_ASYNC_TO_LDS_B128 (ASYNCcnt).
// Each wave computes a 32(m) x 32(n) tile = 4 WMMA accumulators.
// grid = (Nn/32, M/256, batch), K must be 512.
// ---------------------------------------------------------------------------
__global__ __launch_bounds__(256) void gemm16_nt_kernel(
    const _Float16* __restrict__ A, long long aStride, int lda,
    const _Float16* __restrict__ B, long long bStride, int ldb,
    _Float16* __restrict__ outH, float* __restrict__ outF,
    long long oStride, int ldo,
    const float* __restrict__ biasM, const float* __restrict__ biasN,
    const float* __restrict__ resid, long long rStride,
    int K) {
    __shared__ _Float16 sB[32 * SB_LD];  // 32 rows x 512 halves (padded)

    int lane = threadIdx.x & 31;
    int wave = threadIdx.x >> 5;
    int z = blockIdx.z;
    int n_base = blockIdx.x * 32;
    int m_base = (blockIdx.y * 8 + wave) * 32;

    // ---- async stage of the shared B slab: 32 rows x 512 halves = 32 KB ----
    {
        const _Float16* Bb = B + (size_t)z * bStride;
#pragma unroll
        for (int i = 0; i < 8; ++i) {
            int linear = threadIdx.x + i * 256;   // 2048 x b128 transfers total
            int row = linear >> 6;                // 64 x 16B chunks per row
            int chunk = linear & 63;
            const _Float16* g = Bb + (size_t)(n_base + row) * ldb + chunk * 8;
            unsigned ldsb = (unsigned)(size_t)&sB[row * SB_LD + chunk * 8];
            asm volatile("global_load_async_to_lds_b128 %0, %1, off"
                         :: "v"(ldsb), "v"(g) : "memory");
        }
        asm volatile("s_wait_asynccnt 0x0" ::: "memory");
    }
    __syncthreads();

    const _Float16* arow0 = A + (size_t)z * aStride + (size_t)(m_base + (lane & 15)) * lda;
    const _Float16* arow1 = arow0 + (size_t)16 * lda;
    const _Float16* brow0 = sB + (lane & 15) * SB_LD;
    const _Float16* brow1 = brow0 + 16 * SB_LD;

    vf8 acc[4] = {{}, {}, {}, {}};  // [mi*2+ni]
    for (int k0 = 0; k0 < K; k0 += 32) {
        vh16 a0 = frag_load(arow0, k0, lane);
        vh16 a1 = frag_load(arow1, k0, lane);
        vh16 b0 = frag_load(brow0, k0, lane);
        vh16 b1 = frag_load(brow1, k0, lane);
        acc[0] = wmma_acc(a0, b0, acc[0]);
        acc[1] = wmma_acc(a0, b1, acc[1]);
        acc[2] = wmma_acc(a1, b0, acc[2]);
        acc[3] = wmma_acc(a1, b1, acc[3]);
    }

    int nloc = lane & 15;
    int m0 = ((lane >> 4) & 1) * 8;
#pragma unroll
    for (int ti = 0; ti < 4; ++ti) {
        int m_b = m_base + (ti >> 1) * 16;
        int n_b = n_base + (ti & 1) * 16;
        float bn = biasN ? biasN[n_b + nloc] : 0.0f;
        float vals[8];
#pragma unroll
        for (int e = 0; e < 8; ++e)
            vals[e] = acc[ti][e] + bn + (biasM ? biasM[m_b + m0 + e] : 0.0f);

        size_t obase = (size_t)z * oStride + (size_t)(n_b + nloc) * ldo + m_b + m0;
        if (resid) {
            const float* r = resid + (size_t)z * rStride + (size_t)(n_b + nloc) * ldo + m_b + m0;
            vf4 r0 = *(const vf4*)(r);
            vf4 r1 = *(const vf4*)(r + 4);
#pragma unroll
            for (int e = 0; e < 4; ++e) { vals[e] += r0[e]; vals[e + 4] += r1[e]; }
        }
        if (outH) {
            vh8 o;
#pragma unroll
            for (int e = 0; e < 8; ++e) o[e] = (_Float16)vals[e];
            *(vh8*)(outH + obase) = o;
        } else {
            vf4 o0, o1;
#pragma unroll
            for (int e = 0; e < 4; ++e) { o0[e] = vals[e]; o1[e] = vals[e + 4]; }
            *(vf4*)(outF + obase) = o0;
            *(vf4*)(outF + obase + 4) = o1;
        }
    }
}

// ---------------------------------------------------------------------------
// K4: attention. One block per (batch, 16-query tile). 16x4096 f16 score strip
// in LDS (128 KB of the 320 KB WGP LDS).
// Phase 1: Q fragments hoisted to registers once; 8 waves split the 256 key
//          tiles; S = qT x kT via WMMA (+ prefetch of the wave's next K tile).
// Phase 2: exact softmax per row (16-thread groups, shfl_xor width 16).
// Phase 3: 8 waves split channels (64 each); attnT = P x V via WMMA,
//          B-fragments streamed from LDS, A-fragments from v[C,N].
// ---------------------------------------------------------------------------
__global__ __launch_bounds__(256) void attn_kernel(const _Float16* __restrict__ qT,
                                                   const _Float16* __restrict__ kT,
                                                   const _Float16* __restrict__ vM,
                                                   _Float16* __restrict__ attnT) {
    __shared__ _Float16 sP[16 * NSEQ];  // 128 KB score/probability strip

    int b = blockIdx.y;
    int i_base = blockIdx.x * 16;
    const _Float16* q  = qT + (size_t)b * NSEQ * C_CH;
    const _Float16* kk = kT + (size_t)b * NSEQ * C_CH;
    const _Float16* vv = vM + (size_t)b * C_CH * NSEQ;
    _Float16* oT = attnT + (size_t)b * NSEQ * C_CH;

    int lane = threadIdx.x & 31;
    int wave = threadIdx.x >> 5;
    int nloc = lane & 15;
    int m0 = ((lane >> 4) & 1) * 8;
    const float scale = 0.04419417382415922f;  // 1/sqrt(512)

    // ---- Phase 1: scores ----
    {
        const _Float16* qrow = q + (size_t)(i_base + (lane & 15)) * C_CH;
        vh16 qf[16];
#pragma unroll
        for (int s = 0; s < 16; ++s) qf[s] = frag_load(qrow, s * 32, lane);

        for (int jt = wave; jt < NSEQ / 16; jt += 8) {
            int j_base = jt * 16;
            const _Float16* krow = kk + (size_t)(j_base + (lane & 15)) * C_CH;
            if (jt + 8 < NSEQ / 16)
                __builtin_prefetch(krow + (size_t)128 * C_CH, 0, 0);
            vf8 acc = {};
#pragma unroll
            for (int s = 0; s < 16; ++s)
                acc = wmma_acc(qf[s], frag_load(krow, s * 32, lane), acc);
#pragma unroll
            for (int e = 0; e < 8; ++e)
                sP[(m0 + e) * NSEQ + j_base + nloc] = (_Float16)(acc[e] * scale);
        }
    }
    __syncthreads();

    // ---- Phase 2: exact softmax, one row per 16-thread group ----
    {
        int row = threadIdx.x >> 4;
        int col = threadIdx.x & 15;
        _Float16* prow = sP + row * NSEQ;
        float mx = -1e30f;
        for (int j = col; j < NSEQ; j += 16) mx = fmaxf(mx, (float)prow[j]);
        for (int off = 8; off > 0; off >>= 1) mx = fmaxf(mx, __shfl_xor(mx, off, 16));
        float sum = 0.0f;
        for (int j = col; j < NSEQ; j += 16) sum += __expf((float)prow[j] - mx);
        for (int off = 8; off > 0; off >>= 1) sum += __shfl_xor(sum, off, 16);
        float inv = 1.0f / sum;
        for (int j = col; j < NSEQ; j += 16)
            prow[j] = (_Float16)(__expf((float)prow[j] - mx) * inv);
    }
    __syncthreads();

    // ---- Phase 3: attnT[i, c] = sum_j P[i,j] * v[c,j]  (M=c, N=i, K=j) ----
    int c_base = wave * 64;
    const _Float16* prow = sP + (size_t)(lane & 15) * NSEQ;
    const _Float16* v0 = vv + (size_t)(c_base +  0 + (lane & 15)) * NSEQ;
    const _Float16* v1 = vv + (size_t)(c_base + 16 + (lane & 15)) * NSEQ;
    const _Float16* v2 = vv + (size_t)(c_base + 32 + (lane & 15)) * NSEQ;
    const _Float16* v3 = vv + (size_t)(c_base + 48 + (lane & 15)) * NSEQ;
    vf8 a0 = {}, a1 = {}, a2 = {}, a3 = {};
    for (int j0 = 0; j0 < NSEQ; j0 += 32) {
        vh16 bf = frag_load(prow, j0, lane);  // from LDS
        a0 = wmma_acc(frag_load(v0, j0, lane), bf, a0);
        a1 = wmma_acc(frag_load(v1, j0, lane), bf, a1);
        a2 = wmma_acc(frag_load(v2, j0, lane), bf, a2);
        a3 = wmma_acc(frag_load(v3, j0, lane), bf, a3);
    }
    _Float16* obase = oT + (size_t)(i_base + nloc) * C_CH + c_base + m0;
    vh8 o;
#pragma unroll
    for (int e = 0; e < 8; ++e) o[e] = (_Float16)a0[e];
    *(vh8*)(obase) = o;
#pragma unroll
    for (int e = 0; e < 8; ++e) o[e] = (_Float16)a1[e];
    *(vh8*)(obase + 16) = o;
#pragma unroll
    for (int e = 0; e < 8; ++e) o[e] = (_Float16)a2[e];
    *(vh8*)(obase + 32) = o;
#pragma unroll
    for (int e = 0; e < 8; ++e) o[e] = (_Float16)a3[e];
    *(vh8*)(obase + 48) = o;
}

// ---------------------------------------------------------------------------
// Host launch
// ---------------------------------------------------------------------------
extern "C" void kernel_launch(void* const* d_in, const int* in_sizes, int n_in,
                              void* d_out, int out_size, void* d_ws, size_t ws_size,
                              hipStream_t stream) {
    const float* x        = (const float*)d_in[0];
    const float* gn_scale = (const float*)d_in[1];
    const float* gn_bias  = (const float*)d_in[2];
    const float* wq = (const float*)d_in[3];
    const float* bq = (const float*)d_in[4];
    const float* wk = (const float*)d_in[5];
    const float* bk = (const float*)d_in[6];
    const float* wv = (const float*)d_in[7];
    const float* bv = (const float*)d_in[8];
    const float* wo = (const float*)d_in[9];
    const float* bo = (const float*)d_in[10];
    float* out = (float*)d_out;

    const size_t WELEM = (size_t)C_CH * C_CH;           // 262144
    const size_t BNC   = (size_t)NBATCH * NSEQ * C_CH;  // 8388608

    // workspace carve-up (~120 MB)
    _Float16* wqh  = (_Float16*)d_ws;
    _Float16* wkh  = wqh + WELEM;
    _Float16* wvh  = wkh + WELEM;
    _Float16* woh  = wvh + WELEM;
    float*    xn   = (float*)(woh + WELEM);
    _Float16* xnT  = (_Float16*)(xn + BNC);
    _Float16* qT   = xnT + BNC;
    _Float16* kT   = qT + BNC;
    _Float16* vbuf = kT + BNC;
    _Float16* aT   = vbuf + BNC;

    // weight conversion
    {
        dim3 g((unsigned)(WELEM / 4 / 256));
        cvt_f16_kernel<<<g, 256, 0, stream>>>(wq, wqh, (int)WELEM);
        cvt_f16_kernel<<<g, 256, 0, stream>>>(wk, wkh, (int)WELEM);
        cvt_f16_kernel<<<g, 256, 0, stream>>>(wv, wvh, (int)WELEM);
        cvt_f16_kernel<<<g, 256, 0, stream>>>(wo, woh, (int)WELEM);
    }
    // group norm
    gn_kernel<<<dim3(NBATCH * GROUPS), 256, 0, stream>>>(x, gn_scale, gn_bias, xn);
    // transpose+convert -> xnT [B,N,C] f16
    transpose_cvt_kernel<<<dim3(NSEQ / 32, C_CH / 32, NBATCH), 256, 0, stream>>>(xn, xnT);

    const long long sNC = (long long)NSEQ * C_CH;  // per-batch elems (either layout)

    // q projection: M=o(512), N=spatial(4096), K=c -> qT [N,C]
    gemm16_nt_kernel<<<dim3(NSEQ / 32, C_CH / 256, NBATCH), 256, 0, stream>>>(
        wqh, 0, C_CH, xnT, sNC, C_CH, qT, nullptr, sNC, C_CH, bq, nullptr, nullptr, 0, C_CH);
    // k projection -> kT [N,C]
    gemm16_nt_kernel<<<dim3(NSEQ / 32, C_CH / 256, NBATCH), 256, 0, stream>>>(
        wkh, 0, C_CH, xnT, sNC, C_CH, kT, nullptr, sNC, C_CH, bk, nullptr, nullptr, 0, C_CH);
    // v projection: M=spatial, N=o -> natural store gives v [C,N]
    gemm16_nt_kernel<<<dim3(C_CH / 32, NSEQ / 256, NBATCH), 256, 0, stream>>>(
        xnT, sNC, C_CH, wvh, 0, C_CH, vbuf, nullptr, sNC, NSEQ, nullptr, bv, nullptr, 0, C_CH);

    // attention -> attnT [B,N,C] f16
    attn_kernel<<<dim3(NSEQ / 16, NBATCH), 256, 0, stream>>>(qT, kT, vbuf, aT);

    // output projection + bias + residual(xn), f32 -> d_out [B,C,N]
    gemm16_nt_kernel<<<dim3(C_CH / 32, NSEQ / 256, NBATCH), 256, 0, stream>>>(
        aT, sNC, C_CH, woh, 0, C_CH, nullptr, out, sNC, NSEQ, nullptr, bo, xn, sNC, C_CH);
}
